// ContextOnlyModel_3204045603593
// MI455X (gfx1250) — matrix operand order repaired and checked by reading, not compiled, not run on hardware
//
#include <hip/hip_runtime.h>
#include <math.h>

#define HID   64
#define HID2  128
#define NVOC  64
#define NB    2048
#define NL    512
#define NSLOT 8
#define LNEPS 1e-5f

typedef __attribute__((ext_vector_type(16))) _Float16 v16h;
typedef __attribute__((ext_vector_type(8)))  float    v8f;

// ---- WMMA operand gathers per CDNA5 wave32 layouts (CDNA5 ISA §7.12.2) ----
// A (16x32 f16): lane l holds row M = l&15; k-halves: lanes 0-15 -> k 0-7 & 16-23,
// lanes 16-31 -> k 8-15 & 24-31 (2 halves per VGPR, j = packed index).
__device__ __forceinline__ v16h loadA_f32(const float* src, int ld, int row0, int k0, int lane) {
  int m  = row0 + (lane & 15);
  int kb = k0 + ((lane >> 4) << 3);
  const float* p = src + m * ld + kb;
  v16h a;
#pragma unroll
  for (int j = 0; j < 8; ++j) a[j] = (_Float16)p[j];
#pragma unroll
  for (int j = 0; j < 8; ++j) a[8 + j] = (_Float16)p[16 + j];
  return a;
}

__device__ __forceinline__ v16h loadA_f16(const _Float16* src, int ld, int row0, int k0, int lane) {
  int m  = row0 + (lane & 15);
  int kb = k0 + ((lane >> 4) << 3);
  const _Float16* p = src + m * ld + kb;
  v16h a;
#pragma unroll
  for (int j = 0; j < 8; ++j) a[j] = p[j];
#pragma unroll
  for (int j = 0; j < 8; ++j) a[8 + j] = p[16 + j];
  return a;
}

// B (32x16 f16): lane l holds col N = l&15; lanes 0-15 k 0-7 & 16-23, lanes 16-31 k 8-15 & 24-31.
__device__ __forceinline__ v16h loadB_f32(const float* w, int ld, int k0, int n0, int lane) {
  int n  = n0 + (lane & 15);
  int kb = k0 + ((lane >> 4) << 3);
  v16h b;
#pragma unroll
  for (int j = 0; j < 16; ++j) b[j] = (_Float16)w[(kb + (j < 8 ? j : j + 8)) * ld + n];
  return b;
}

// B = Q^T: element (k, n) = Q[n][k]
__device__ __forceinline__ v16h loadBT_f32(const float* q, int ld, int k0, int n0, int lane) {
  int n  = n0 + (lane & 15);
  int kb = k0 + ((lane >> 4) << 3);
  const float* p = q + n * ld + kb;
  v16h b;
#pragma unroll
  for (int j = 0; j < 16; ++j) b[j] = (_Float16)p[j < 8 ? j : j + 8];
  return b;
}

#define WMMA_F16(A, B, C) \
  __builtin_amdgcn_wmma_f32_16x16x32_f16(false, (A), false, (B), (short)0, (C), false, false)

// ============================================================================
// Kernel 1: build all per-vocab tables. One block, 128 threads = 4 waves;
// wave w owns vocab rows 16w..16w+15 (M-tile). h depends only on vocab id.
// ============================================================================
__global__ __launch_bounds__(128) void build_tables(
    const float* __restrict__ embed, const float* __restrict__ W1, const float* __restrict__ b1,
    const float* __restrict__ W2, const float* __restrict__ b2,
    const float* __restrict__ ln_g, const float* __restrict__ ln_b,
    const float* __restrict__ gW, const float* __restrict__ gb,
    const float* __restrict__ qW, const float* __restrict__ qb,
    const float* __restrict__ oW, const float* __restrict__ ob,
    float* __restrict__ Stab, float* __restrict__ Otab, int* __restrict__ perm) {
  __shared__ _Float16 z1[4][16][HID2];   // 16 KB  relu(h0@W1+b1), per-wave tile
  __shared__ float    htab[NVOC][HID];   // 16 KB  LayerNormed hidden per vocab id
  __shared__ float    qtab[NVOC][HID];   // 16 KB  htab @ qW + qb
  __shared__ float    gtab[NVOC];        // gate per vocab id

  const int tid  = threadIdx.x;
  const int w    = tid >> 5;
  const int lane = tid & 31;
  const int M0   = w << 4;
  const int ncol = lane & 15;            // C-layout: lane's column within n-tile
  const int rb   = (lane >> 4) << 3;     // C-layout: lane's row base within m-tile

  // ---- Phase 1: FFN.  h0 tile IS the embed table (tokens == vocab ids) ----
  v16h a0 = loadA_f32(embed, HID, M0, 0, lane);
  v16h a1 = loadA_f32(embed, HID, M0, 32, lane);
#pragma unroll
  for (int nt = 0; nt < 8; ++nt) {       // layer1: N=128 -> 8 n-tiles, K=64 -> 2 steps
    v8f c = {};
    c = WMMA_F16(a0, loadB_f32(W1, HID2, 0, nt * 16, lane), c);
    c = WMMA_F16(a1, loadB_f32(W1, HID2, 32, nt * 16, lane), c);
    int n = nt * 16 + ncol;
    float bias = b1[n];
#pragma unroll
    for (int i = 0; i < 8; ++i)
      z1[w][rb + i][n] = (_Float16)fmaxf(c[i] + bias, 0.f);
  }
  __syncthreads();

  v16h a2[4];
#pragma unroll
  for (int kc = 0; kc < 4; ++kc) a2[kc] = loadA_f16(&z1[w][0][0], HID2, 0, kc * 32, lane);
#pragma unroll
  for (int nt = 0; nt < 4; ++nt) {       // layer2: N=64 -> 4 n-tiles, K=128 -> 4 steps
    v8f c = {};
#pragma unroll
    for (int kc = 0; kc < 4; ++kc)
      c = WMMA_F16(a2[kc], loadB_f32(W2, HID, kc * 32, nt * 16, lane), c);
    int n = nt * 16 + ncol;
    float bias = b2[n];
#pragma unroll
    for (int i = 0; i < 8; ++i) {
      int v = M0 + rb + i;
      htab[v][n] = embed[v * HID + n] + c[i] + bias;   // residual x = h0 + ff
    }
  }
  __syncthreads();

  // ---- LayerNorm rows + gate: 2 threads per row (partner = tid^1, same wave) ----
  {
    int r  = tid >> 1;
    int c0 = (tid & 1) * 32;
    float s = 0.f, ss = 0.f;
    for (int c = 0; c < 32; ++c) { float x = htab[r][c0 + c]; s += x; ss += x * x; }
    s  += __shfl_xor(s, 1, 32);
    ss += __shfl_xor(ss, 1, 32);
    float mu  = s * (1.f / 64.f);
    float var = ss * (1.f / 64.f) - mu * mu;
    float inv = rsqrtf(var + LNEPS);
    float ga = 0.f;
    for (int c = 0; c < 32; ++c) {
      int cc = c0 + c;
      float hv = (htab[r][cc] - mu) * inv * ln_g[cc] + ln_b[cc];
      htab[r][cc] = hv;
      ga += hv * gW[cc];
    }
    ga += __shfl_xor(ga, 1, 32);
    if ((tid & 1) == 0) gtab[r] = 1.f / (1.f + expf(-(ga + gb[0])));
  }
  __syncthreads();

  // ---- Phase 2: qtab = htab @ qW + qb ----
  v16h ha0 = loadA_f32(&htab[0][0], HID, M0, 0, lane);
  v16h ha1 = loadA_f32(&htab[0][0], HID, M0, 32, lane);
#pragma unroll
  for (int nt = 0; nt < 4; ++nt) {
    v8f c = {};
    c = WMMA_F16(ha0, loadB_f32(qW, HID, 0, nt * 16, lane), c);
    c = WMMA_F16(ha1, loadB_f32(qW, HID, 32, nt * 16, lane), c);
    int n = nt * 16 + ncol;
    float bias = qb[n];
#pragma unroll
    for (int i = 0; i < 8; ++i) qtab[M0 + rb + i][n] = c[i] + bias;
  }
  __syncthreads();

  // ---- Phase 3: Stab[v][u] = htab[v] . qtab[u]  (B = qtab^T) ----
#pragma unroll
  for (int nt = 0; nt < 4; ++nt) {
    v8f c = {};
    c = WMMA_F16(ha0, loadBT_f32(&qtab[0][0], HID, 0, nt * 16, lane), c);
    c = WMMA_F16(ha1, loadBT_f32(&qtab[0][0], HID, 32, nt * 16, lane), c);
    int n = nt * 16 + ncol;
#pragma unroll
    for (int i = 0; i < 8; ++i) Stab[(M0 + rb + i) * HID + n] = c[i];
  }

  // ---- Phase 4: Otab = htab @ oW + ob (ob folded: Σ attn = 1) ----
#pragma unroll
  for (int nt = 0; nt < 4; ++nt) {
    v8f c = {};
    c = WMMA_F16(ha0, loadB_f32(oW, HID, 0, nt * 16, lane), c);
    c = WMMA_F16(ha1, loadB_f32(oW, HID, 32, nt * 16, lane), c);
    int n = nt * 16 + ncol;
    float bias = ob[n];
#pragma unroll
    for (int i = 0; i < 8; ++i) Otab[(M0 + rb + i) * HID + n] = c[i] + bias;
  }

  // ---- Phase 5: perm = vocab ids ranked by gate, descending ----
  if (tid < NVOC) {
    float g = gtab[tid];
    int rank = 0;
    for (int u = 0; u < NVOC; ++u) {
      float gu = gtab[u];
      rank += (gu > g) || (gu == g && u < tid);
    }
    perm[rank] = tid;
  }
}

// ============================================================================
// Kernel 2: per batch row (1 wave/row, 8 rows/block). Top-8 via 64-bin
// histogram + perm walk (gate ties <=> identical tokens <=> identical h,
// and attention is permutation-invariant, so the selected multiset suffices).
// ============================================================================
__global__ __launch_bounds__(256) void decode_rows(
    const int* __restrict__ seq, const float* __restrict__ Stab,
    const float* __restrict__ Otab, const int* __restrict__ perm_g,
    float* __restrict__ out) {
  __shared__ int   permS[NVOC];
  __shared__ int   cnt[8][NVOC];
  __shared__ int   selS[8][NSLOT];
  __shared__ float attnS[8][NSLOT];

  const int tid  = threadIdx.x;
  const int w    = tid >> 5;
  const int lane = tid & 31;
  const int b    = blockIdx.x * 8 + w;

  if (tid < NVOC) permS[tid] = perm_g[tid];
  cnt[w][lane]      = 0;
  cnt[w][lane + 32] = 0;
  __syncthreads();

  const int* row = seq + b * NL;
#pragma unroll
  for (int t = 0; t < 16; ++t) {
    int tok = row[t * 32 + lane];
    atomicAdd(&cnt[w][tok], 1);
  }
  __syncthreads();

  if (lane == 0) {
    int rem = NSLOT, si = 0;
    for (int p = 0; p < NVOC && rem > 0; ++p) {
      int v = permS[p];
      int c = cnt[w][v];
      int take = c < rem ? c : rem;
      for (int t = 0; t < take; ++t) selS[w][si++] = v;
      rem -= take;
    }
    int u = row[NL - 1];
    float sc[NSLOT];
    float m = -1e30f;
    for (int k = 0; k < NSLOT; ++k) {
      sc[k] = Stab[selS[w][k] * HID + u];
      m = fmaxf(m, sc[k]);
    }
    float sum = 0.f;
    for (int k = 0; k < NSLOT; ++k) { sc[k] = expf(sc[k] - m); sum += sc[k]; }
    float inv = 1.f / sum;
    for (int k = 0; k < NSLOT; ++k) attnS[w][k] = sc[k] * inv;
  }
  __syncthreads();

  float acc0 = 0.f, acc1 = 0.f;
#pragma unroll
  for (int k = 0; k < NSLOT; ++k) {
    float a = attnS[w][k];
    const float* orow = Otab + selS[w][k] * HID;
    acc0 += a * orow[lane];
    acc1 += a * orow[lane + 32];
  }
  out[b * HID + lane]      = acc0;
  out[b * HID + lane + 32] = acc1;
}

// ============================================================================
extern "C" void kernel_launch(void* const* d_in, const int* in_sizes, int n_in,
                              void* d_out, int out_size, void* d_ws, size_t ws_size,
                              hipStream_t stream) {
  (void)in_sizes; (void)n_in; (void)out_size; (void)ws_size;
  const int*   seq   = (const int*)d_in[0];
  const float* embed = (const float*)d_in[1];
  const float* W1    = (const float*)d_in[2];
  const float* b1    = (const float*)d_in[3];
  const float* W2    = (const float*)d_in[4];
  const float* b2    = (const float*)d_in[5];
  const float* ln_g  = (const float*)d_in[6];
  const float* ln_b  = (const float*)d_in[7];
  const float* gW    = (const float*)d_in[8];
  const float* gb    = (const float*)d_in[9];
  const float* qW    = (const float*)d_in[10];
  const float* qb    = (const float*)d_in[11];
  const float* oW    = (const float*)d_in[12];
  const float* ob    = (const float*)d_in[13];

  float* Stab = (float*)d_ws;                 // 64*64 f32
  float* Otab = Stab + NVOC * HID;            // 64*64 f32
  int*   perm = (int*)(Otab + NVOC * HID);    // 64 int

  build_tables<<<1, 128, 0, stream>>>(embed, W1, b1, W2, b2, ln_g, ln_b,
                                      gW, gb, qW, qb, oW, ob, Stab, Otab, perm);
  decode_rows<<<NB / 8, 256, 0, stream>>>(seq, Stab, Otab, perm, (float*)d_out);
}
